// TriSourceQueryRouter_70841190580295
// MI455X (gfx1250) — matrix operand order repaired and checked by reading, not compiled, not run on hardware
//
#include <hip/hip_runtime.h>
#include <hip/hip_bf16.h>

// ---------------------------------------------------------------------------
// TriSourceQueryRouter for MI455X (gfx1250, wave32, WMMA, async-LDS)
//   B=16, N=7168 (4096 lidar + 2048 prop + 1024 glob), D=512, K=900
// ---------------------------------------------------------------------------

typedef __attribute__((ext_vector_type(16))) __bf16 v16bf;
typedef __attribute__((ext_vector_type(8)))  float  v8f;

#define BB      16
#define NTOT    7168
#define NL      4096
#define NP      2048
#define NG      1024
#define DD      512
#define KK      900

// output offsets (floats), outputs concatenated flat in return order
#define OFF_REFS    (BB*KK*DD)          // 7372800
#define OFF_SCORES  (OFF_REFS + BB*KK*3)
#define OFF_SRC     (OFF_SCORES + BB*KK)
#define OFF_LOGITS  (OFF_SRC + BB*KK)

// workspace offsets (bytes)
#define WS_PACKW1   0                         // 512*512 bf16 = 524288 B
#define WS_KSCORE   (512*512*2)               // B*N f32 = 458752 B
#define WS_KLOGIT   (WS_KSCORE + BB*NTOT*4)
#define WS_TOPIDX   (WS_KLOGIT + BB*NTOT*4)   // B*K int

__device__ __forceinline__ unsigned short f2bf(float x) {
    unsigned u = __float_as_uint(x);
    unsigned r = u + 0x7FFFu + ((u >> 16) & 1u);   // round-to-nearest-even
    return (unsigned short)(r >> 16);
}

__device__ __forceinline__ unsigned mapKey(float x) {
    // monotone float->uint: larger float -> larger uint
    unsigned u = __float_as_uint(x);
    return (u & 0x80000000u) ? ~u : (u | 0x80000000u);
}

// CDNA5 async memory->LDS copy of 16 bytes (per active lane), ASYNCcnt-tracked.
// LDS address = low 32 bits of the flat shared-aperture pointer (ISA 10.2).
__device__ __forceinline__ void async_copy16(const void* gsrc, void* lds_dst) {
    unsigned laddr = (unsigned)(uintptr_t)lds_dst;
    asm volatile("global_load_async_to_lds_b128 %0, %1, off"
                 :: "v"(laddr), "v"(gsrc) : "memory");
}
__device__ __forceinline__ void wait_async0() {
    asm volatile("s_wait_asynccnt 0x0" ::: "memory");
}

// ---------------------------------------------------------------------------
// Kernel 0: pack w1[0:512, 0:512] (f32, row-major [K][N]) into bf16 WMMA
// B-fragments. Per 16-col output tile n: one contiguous 16 KB chunk
// (async-DMA friendly). Within a (n,kk) fragment (1 KB), the 16 halfs of each
// lane are split into two 16 B half-fragments stored lane-major at stride 16
// (bank-conflict-free ds_load_b128):
//   bytes [l*16 .. l*16+16)        : lane l, halfs e=0..7
//   bytes [512 + l*16 .. +16)      : lane l, halfs e=8..15
// Fragment element (lane l, e) = w1[K = kk*32 + (l/16)*8 + (e%8) + (e/8)*16][N = n*16 + l%16].
// ---------------------------------------------------------------------------
__global__ void pack_w1_kernel(const float* __restrict__ w1,
                               unsigned short* __restrict__ packed) {
    int t = blockIdx.x * blockDim.x + threadIdx.x;   // 0 .. 262143 (halfs)
    int n  = t >> 13;
    int kk = (t >> 9) & 15;
    int within = t & 511;
    int half = within >> 8;         // 0: e=0..7 piece, 1: e=8..15 piece
    int rem = within & 255;
    int l = rem >> 3;
    int e = (rem & 7) + half * 8;
    int K = kk * 32 + (l >> 4) * 8 + (e & 7) + half * 16;
    int N = n * 16 + (l & 15);
    packed[t] = f2bf(w1[K * DD + N]);
}

// ---------------------------------------------------------------------------
// Kernel 1: fused MLP. Each wave owns TWO 16-row slabs (32 rows) of the
// flattened [B*N] row space (32-aligned groups never cross source/batch
// boundaries). bf16 WMMA over K=512; the 513th feature (score) enters as an
// exact f32 rank-1 epilogue term. The 4 waves of a block share B tiles staged
// in LDS by double-buffered async copies (ASYNCcnt). Mt=2 register blocking:
// each B fragment read from LDS feeds two wmma (halves LDS feed per wmma to
// ~256 B/clk per WGP = LDS peak), with two independent accumulator chains.
// ---------------------------------------------------------------------------
__global__ void __launch_bounds__(128, 1)
mlp_score_kernel(const float* __restrict__ lq, const float* __restrict__ ls,
                 const float* __restrict__ pq, const float* __restrict__ ps,
                 const float* __restrict__ gq, const float* __restrict__ gs,
                 const float* __restrict__ emb,
                 const float* __restrict__ w1, const float* __restrict__ b1,
                 const float* __restrict__ w2, const float* __restrict__ b2,
                 const unsigned short* __restrict__ packedW1,
                 float* __restrict__ keep_scores,
                 float* __restrict__ keep_logits) {
    __shared__ __align__(32) unsigned char bbuf[2][16384];   // double-buffered B tile

    const int tid  = threadIdx.x;
    const int lane = tid & 31;
    const int wave = tid >> 5;
    const int slab = blockIdx.x * 4 + wave;
    const int rowbase = slab * 32;                 // flattened row base (32 rows)
    const int b  = rowbase / NTOT;
    const int n0 = rowbase - b * NTOT;

    const char* wbytes = (const char*)packedW1;

    // kick off async DMA of B tile 0 while we build A fragments
    {
        const char* src = wbytes + (size_t)tid * 128;
        char* dst = (char*)&bbuf[0][tid * 128];
#pragma unroll
        for (int i = 0; i < 8; ++i) async_copy16(src + i * 16, dst + i * 16);
    }

    const float* qbase; const float* sbase; int src, local0;
    if (n0 < NL)            { src = 0; local0 = b * NL + n0;             qbase = lq; sbase = ls; }
    else if (n0 < NL + NP)  { src = 1; local0 = b * NP + (n0 - NL);      qbase = pq; sbase = ps; }
    else                    { src = 2; local0 = b * NG + (n0 - NL - NP); qbase = gq; sbase = gs; }

    // ---- build A fragments for both 16-row tiles: 16x32 bf16, ISA layout ----
    // lane<16: M=lane,   halfs e=0..7 -> K=kk*32+0..7,  e=8..15 -> K=kk*32+16..23
    // lane>=16:M=lane-16,halfs e=0..7 -> K=kk*32+8..15, e=8..15 -> K=kk*32+24..31
    const float* eptr = emb + src * DD;
    const int kb = (lane >> 4) * 8;

    v16bf afrag[2][16];
#pragma unroll
    for (int t = 0; t < 2; ++t) {
        const float* arow = qbase + (size_t)(local0 + t * 16 + (lane & 15)) * DD;
#pragma unroll
        for (int kk = 0; kk < 16; ++kk) {
            const int kbase = kk * 32 + kb;
            float4 f0 = *(const float4*)(arow + kbase);
            float4 f1 = *(const float4*)(arow + kbase + 4);
            float4 f2 = *(const float4*)(arow + kbase + 16);
            float4 f3 = *(const float4*)(arow + kbase + 20);
            float4 e0 = *(const float4*)(eptr + kbase);
            float4 e1 = *(const float4*)(eptr + kbase + 4);
            float4 e2 = *(const float4*)(eptr + kbase + 16);
            float4 e3 = *(const float4*)(eptr + kbase + 20);
            union { v16bf v; unsigned short u[16]; } tmp;
            tmp.u[0]  = f2bf(f0.x + e0.x); tmp.u[1]  = f2bf(f0.y + e0.y);
            tmp.u[2]  = f2bf(f0.z + e0.z); tmp.u[3]  = f2bf(f0.w + e0.w);
            tmp.u[4]  = f2bf(f1.x + e1.x); tmp.u[5]  = f2bf(f1.y + e1.y);
            tmp.u[6]  = f2bf(f1.z + e1.z); tmp.u[7]  = f2bf(f1.w + e1.w);
            tmp.u[8]  = f2bf(f2.x + e2.x); tmp.u[9]  = f2bf(f2.y + e2.y);
            tmp.u[10] = f2bf(f2.z + e2.z); tmp.u[11] = f2bf(f2.w + e2.w);
            tmp.u[12] = f2bf(f3.x + e3.x); tmp.u[13] = f2bf(f3.y + e3.y);
            tmp.u[14] = f2bf(f3.z + e3.z); tmp.u[15] = f2bf(f3.w + e3.w);
            afrag[t][kk] = tmp.v;
        }
    }

    // scores for this lane's rows (C-layout: row M = j + 8*(lane/16))
    float scoreReg[2][8];
    const float* scp = sbase + local0;
#pragma unroll
    for (int t = 0; t < 2; ++t)
#pragma unroll
        for (int j = 0; j < 8; ++j)
            scoreReg[t][j] = scp[t * 16 + j + ((lane >> 4) << 3)];

    const float* w1last = w1 + DD * DD;            // row 512 of w1 (score row)
    float logitAcc[2][8];
#pragma unroll
    for (int t = 0; t < 2; ++t)
#pragma unroll
        for (int j = 0; j < 8; ++j) logitAcc[t][j] = 0.f;

    // B tile 0 must be resident before compute
    wait_async0();
    __syncthreads();

#pragma unroll 1
    for (int n = 0; n < 32; ++n) {
        const int cur = n & 1;
        // prefetch next B tile into the other buffer (uniform branch)
        if (n < 31) {
            const char* psrc = wbytes + (size_t)(n + 1) * 16384 + (size_t)tid * 128;
            char* pdst = (char*)&bbuf[cur ^ 1][tid * 128];
#pragma unroll
            for (int i = 0; i < 8; ++i) async_copy16(psrc + i * 16, pdst + i * 16);
        }

        const uint4* pB4 = (const uint4*)&bbuf[cur][0];
        v8f acc0 = {0.f,0.f,0.f,0.f,0.f,0.f,0.f,0.f};
        v8f acc1 = {0.f,0.f,0.f,0.f,0.f,0.f,0.f,0.f};
#pragma unroll
        for (int kk = 0; kk < 16; ++kk) {
            // conflict-free: two stride-16 half-fragment ds_load_b128
            union { v16bf v; uint4 q[2]; } bu;
            bu.q[0] = pB4[kk * 64 + lane];
            bu.q[1] = pB4[kk * 64 + 32 + lane];
            acc0 = __builtin_amdgcn_wmma_f32_16x16x32_bf16(
                false, afrag[0][kk], false, bu.v, (short)0, acc0, false, false);
            acc1 = __builtin_amdgcn_wmma_f32_16x16x32_bf16(
                false, afrag[1][kk], false, bu.v, (short)0, acc1, false, false);
        }
        // epilogue for 16 cols of this tile: lane holds col = n*16 + lane%16,
        // rows M = j + 8*(lane/16)
        const int col = n * 16 + (lane & 15);
        const float b1c = b1[col];
        const float wl  = w1last[col];
        const float w2c = w2[col];
#pragma unroll
        for (int j = 0; j < 8; ++j) {
            float h0 = acc0[j] + b1c + scoreReg[0][j] * wl;
            h0 = h0 > 0.f ? h0 : 0.f;              // ReLU
            logitAcc[0][j] = fmaf(h0, w2c, logitAcc[0][j]);
            float h1 = acc1[j] + b1c + scoreReg[1][j] * wl;
            h1 = h1 > 0.f ? h1 : 0.f;
            logitAcc[1][j] = fmaf(h1, w2c, logitAcc[1][j]);
        }

        // next buffer ready + all waves done reading current before reuse
        wait_async0();
        __syncthreads();
    }

    // reduce across the 16 lanes holding different cols of the same rows
#pragma unroll
    for (int t = 0; t < 2; ++t)
#pragma unroll
        for (int j = 0; j < 8; ++j) {
            float v = logitAcc[t][j];
            v += __shfl_xor(v, 1, 16);
            v += __shfl_xor(v, 2, 16);
            v += __shfl_xor(v, 4, 16);
            v += __shfl_xor(v, 8, 16);
            logitAcc[t][j] = v;
        }
    const float b2v = b2[0];
    if ((lane & 15) == 0) {
        const int half = lane >> 4;
#pragma unroll
        for (int t = 0; t < 2; ++t)
#pragma unroll
            for (int j = 0; j < 8; ++j) {
                const int m = t * 16 + j + half * 8;
                const float lg = logitAcc[t][j] + b2v;
                keep_logits[rowbase + m] = lg;
                keep_scores[rowbase + m] = lg + scoreReg[t][j];
            }
    }
}

// ---------------------------------------------------------------------------
// Kernel 2: per-batch exact top-900 of keep_scores[b, 0:7168].
// 4-round MSB radix select -> exact threshold; deterministic ordered
// compaction; 1024-wide bitonic sort (key desc, idx asc) in LDS.
// ---------------------------------------------------------------------------
__global__ void __launch_bounds__(256)
topk_kernel(const float* __restrict__ keep_scores, int* __restrict__ top_idx) {
    const int b   = blockIdx.x;
    const int tid = threadIdx.x;
    const float* ks = keep_scores + b * NTOT;

    __shared__ unsigned hist[256];
    __shared__ int scanBuf[256];
    __shared__ unsigned selKey[1024];
    __shared__ int selIdx[1024];
    __shared__ unsigned s_prefix, s_prefmask;
    __shared__ int s_need, s_base;

    if (tid == 0) { s_prefix = 0u; s_prefmask = 0u; s_need = KK; }
    __syncthreads();

    for (int round = 0; round < 4; ++round) {
        const int shift = 24 - round * 8;
        hist[tid] = 0u;
        __syncthreads();
        const unsigned pfx = s_prefix, msk = s_prefmask;
        for (int i = tid; i < NTOT; i += 256) {
            unsigned u = mapKey(ks[i]);
            if ((u & msk) == pfx) atomicAdd(&hist[(u >> shift) & 255u], 1u);
        }
        __syncthreads();
        if (tid == 0) {
            int need = s_need, cum = 0, bin = 255;
            for (; bin >= 0; --bin) {
                int c = (int)hist[bin];
                if (cum + c >= need) break;
                cum += c;
            }
            if (bin < 0) bin = 0;                  // safety
            s_need = need - cum;
            s_prefix = pfx | ((unsigned)bin << shift);
            s_prefmask = msk | (255u << shift);
        }
        __syncthreads();
    }
    const unsigned T = s_prefix;                   // exact mapped threshold

    if (tid == 0) s_base = 0;
    __syncthreads();
    for (int pass = 0; pass < 2; ++pass) {         // pass0: >T, pass1: ==T
        for (int start = 0; start < NTOT; start += 256) {
            const int i = start + tid;
            int flag = 0; unsigned u = 0u;
            if (i < NTOT) {
                u = mapKey(ks[i]);
                flag = pass == 0 ? (u > T) : (u == T);
            }
            scanBuf[tid] = flag;
            __syncthreads();
            for (int off = 1; off < 256; off <<= 1) {
                int v = (tid >= off) ? scanBuf[tid - off] : 0;
                __syncthreads();
                scanBuf[tid] += v;
                __syncthreads();
            }
            if (flag) {
                int pos = s_base + scanBuf[tid] - 1;
                if (pos < KK) { selKey[pos] = u; selIdx[pos] = i; }
            }
            __syncthreads();
            if (tid == 0) s_base += scanBuf[255];
            __syncthreads();
        }
    }

    // pad 900..1023 so padding sinks to the bottom of the sort
    for (int i = tid; i < 1024; i += 256)
        if (i >= KK) { selKey[i] = 0u; selIdx[i] = 0x7FFFFFFF; }
    __syncthreads();

    // bitonic sort: key descending, idx ascending on ties
    for (int size = 2; size <= 1024; size <<= 1) {
        for (int stride = size >> 1; stride > 0; stride >>= 1) {
            __syncthreads();
            for (int i = tid; i < 1024; i += 256) {
                const int j = i ^ stride;
                if (j > i) {
                    const bool descBlock = ((i & size) == 0);
                    unsigned k1 = selKey[i], k2 = selKey[j];
                    int i1 = selIdx[i], i2 = selIdx[j];
                    const bool firstRanks = (k1 > k2) || (k1 == k2 && i1 < i2);
                    const bool doSwap = descBlock ? !firstRanks : firstRanks;
                    if (doSwap) {
                        selKey[i] = k2; selKey[j] = k1;
                        selIdx[i] = i2; selIdx[j] = i1;
                    }
                }
            }
        }
    }
    __syncthreads();
    for (int k = tid; k < KK; k += 256) top_idx[b * KK + k] = selIdx[k];
}

// ---------------------------------------------------------------------------
// Kernel 3: multi-gather; one block per (b,k) output row.
// ---------------------------------------------------------------------------
__global__ void __launch_bounds__(128)
gather_kernel(const float* __restrict__ lq, const float* __restrict__ lr, const float* __restrict__ ls,
              const float* __restrict__ pq, const float* __restrict__ pr, const float* __restrict__ ps,
              const float* __restrict__ gq, const float* __restrict__ gr, const float* __restrict__ gs,
              const float* __restrict__ emb,
              const float* __restrict__ keep_logits,
              const int* __restrict__ top_idx,
              float* __restrict__ out) {
    const int row = blockIdx.x;                    // 0 .. B*K-1
    const int b = row / KK;
    const int idx = top_idx[row];

    int src, local; const float* q; const float* r; const float* s;
    if (idx < NL)           { src = 0; local = b * NL + idx;             q = lq; r = lr; s = ls; }
    else if (idx < NL + NP) { src = 1; local = b * NP + (idx - NL);      q = pq; r = pr; s = ps; }
    else                    { src = 2; local = b * NG + (idx - NL - NP); q = gq; r = gr; s = gs; }

    const float* qrow = q + (size_t)local * DD;
    const float* e = emb + src * DD;
    float* qout = out + (size_t)row * DD;
    const int c = threadIdx.x * 4;                 // 128 threads x 4 cols = 512
    float4 v  = *(const float4*)(qrow + c);
    float4 ev = *(const float4*)(e + c);
    v.x += ev.x; v.y += ev.y; v.z += ev.z; v.w += ev.w;
    *(float4*)(qout + c) = v;

    if (threadIdx.x < 3)
        out[OFF_REFS + row * 3 + threadIdx.x] = r[(size_t)local * 3 + threadIdx.x];
    if (threadIdx.x == 0) {
        out[OFF_SCORES + row] = s[local];
        out[OFF_SRC + row]    = (float)src;
        out[OFF_LOGITS + row] = keep_logits[b * NTOT + idx];
    }
}

// ---------------------------------------------------------------------------
extern "C" void kernel_launch(void* const* d_in, const int* in_sizes, int n_in,
                              void* d_out, int out_size, void* d_ws, size_t ws_size,
                              hipStream_t stream) {
    const float* lq  = (const float*)d_in[0];
    const float* lr  = (const float*)d_in[1];
    const float* ls  = (const float*)d_in[2];
    const float* pq  = (const float*)d_in[3];
    const float* pr  = (const float*)d_in[4];
    const float* ps  = (const float*)d_in[5];
    const float* gq  = (const float*)d_in[6];
    const float* gr  = (const float*)d_in[7];
    const float* gs  = (const float*)d_in[8];
    const float* emb = (const float*)d_in[9];
    const float* w1  = (const float*)d_in[10];
    const float* b1  = (const float*)d_in[11];
    const float* w2  = (const float*)d_in[12];
    const float* b2  = (const float*)d_in[13];

    char* ws = (char*)d_ws;
    unsigned short* packedW1 = (unsigned short*)(ws + WS_PACKW1);
    float* keep_scores = (float*)(ws + WS_KSCORE);
    float* keep_logits = (float*)(ws + WS_KLOGIT);
    int*   top_idx     = (int*)(ws + WS_TOPIDX);
    float* out = (float*)d_out;

    // 0) pack w1 -> bf16 WMMA B-fragments, tile-contiguous, bank-swizzled
    pack_w1_kernel<<<1024, 256, 0, stream>>>(w1, packedW1);
    // 1) fused MLP: 3584 waves x 32 rows, 4 waves per block, B tiles shared
    //    via double-buffered async LDS staging, Mt=2 register blocking
    mlp_score_kernel<<<896, 128, 0, stream>>>(lq, ls, pq, ps, gq, gs, emb,
                                              w1, b1, w2, b2, packedW1,
                                              keep_scores, keep_logits);
    // 2) per-batch exact top-900
    topk_kernel<<<BB, 256, 0, stream>>>(keep_scores, top_idx);
    // 3) multi-gather
    gather_kernel<<<BB * KK, 128, 0, stream>>>(lq, lr, ls, pq, pr, ps,
                                               gq, gr, gs, emb, keep_logits,
                                               top_idx, out);
}
